// ResidualStreamMapping_6640019440347
// MI455X (gfx1250) — matrix operand order repaired and checked by reading, not compiled, not run on hardware
//
#include <hip/hip_runtime.h>
#include <hip/hip_bf16.h>

typedef __attribute__((ext_vector_type(16))) __bf16 v16bf;
typedef __attribute__((ext_vector_type(8)))  float  v8f;
typedef __attribute__((ext_vector_type(4)))  float  f4;

#define D_EMBD     4096
#define KC         256         // K-chunk staged in LDS per iteration
#define NCHUNK     (D_EMBD / KC)
#define ROWSTRIDE  132         // dword (bf16-pair) row stride: 128 + 4 pad -> conflict-free
#define NTHREADS   128         // 4 waves of 32
#define TOK_PER_WG 16
#define WPK_ROW    (D_EMBD / 2)  // packed W row stride in dwords

// True packed convert: one v_cvt_pk_bf16_f32 per bf16 pair (lo -> bits[15:0]).
// The compiler itself emits this mnemonic on gfx1250, so the assembler accepts it.
__device__ __forceinline__ unsigned pack_bf16(float lo, float hi) {
    unsigned r;
    asm("v_cvt_pk_bf16_f32 %0, %1, %2" : "=v"(r) : "v"(lo), "v"(hi));
    return r;
}

// ---- prep: convert W [16,4096] fp32 -> packed bf16 pairs in d_ws (run once per launch) ----
__global__ __launch_bounds__(NTHREADS)
void pack_w_kernel(const float* __restrict__ W, unsigned* __restrict__ wpk) {
    const int r   = blockIdx.x;          // 16 rows
    const int tid = threadIdx.x;
    const float* src = W   + (size_t)r * D_EMBD;
    unsigned*    dst = wpk + (size_t)r * WPK_ROW;
    #pragma unroll
    for (int it = 0; it < D_EMBD / (NTHREADS * 4); ++it) {   // 8 iters
        const int col = (it * NTHREADS + tid) * 4;
        f4 v = *(const f4*)(src + col);
        dst[(col >> 1)]     = pack_bf16(v.x, v.y);
        dst[(col >> 1) + 1] = pack_bf16(v.z, v.w);
    }
}

__global__ __launch_bounds__(NTHREADS)
void rsm_kernel(const float* __restrict__ x,        // [tokens, 4096]
                const unsigned* __restrict__ wpk,   // [16, 2048] packed bf16 pairs
                const float* __restrict__ bias,     // [4,4] flat 16
                const float* __restrict__ alpha_p,
                float* __restrict__ out)            // [tokens, 16]
{
    __shared__ __align__(16) unsigned sA[2][16 * ROWSTRIDE]; // double-buffered x tile
    __shared__ float ssq[16];
    __shared__ float accb[4 * 16 * 16];                      // per-wave partial C tiles

    const int tid  = threadIdx.x;
    const int wave = tid >> 5;
    const int lane = tid & 31;
    const int half = lane >> 4;        // 0: lanes 0-15, 1: lanes 16-31
    const int j    = lane & 15;        // A row (token) and B column (feature)
    const int row  = tid >> 3;         // staging row (0..15), fixed per thread
    const int sub  = tid & 7;          // staging sub-lane within row
    const int tok0 = blockIdx.x * TOK_PER_WG;

    if (tid < 16) ssq[tid] = 0.0f;

    const float* xrow = x + (size_t)(tok0 + row) * D_EMBD;

    float ssq_local = 0.0f;
    v8f acc = {};

    // ---- prologue: stage chunk 0 ----
    #pragma unroll
    for (int it = 0; it < 8; ++it) {
        const int col = it * 32 + sub * 4;
        f4 vx = __builtin_nontemporal_load((const f4*)(xrow + col));
        ssq_local += vx.x * vx.x + vx.y * vx.y + vx.z * vx.z + vx.w * vx.w;
        const int p = col >> 1;
        sA[0][row * ROWSTRIDE + p]     = pack_bf16(vx.x, vx.y);
        sA[0][row * ROWSTRIDE + p + 1] = pack_bf16(vx.z, vx.w);
    }
    __syncthreads();

    for (int c = 0; c < NCHUNK; ++c) {
        const bool more = (c + 1) < NCHUNK;
        // ---- issue next chunk's streaming loads early ----
        f4 rx[8];
        if (more) {
            #pragma unroll
            for (int it = 0; it < 8; ++it) {
                const int col = (c + 1) * KC + it * 32 + sub * 4;
                rx[it] = __builtin_nontemporal_load((const f4*)(xrow + col));
            }
        }
        // ---- compute chunk c: A from LDS, B straight from packed global W (L2-hot) ----
        const unsigned* bufA = sA[c & 1];
        #pragma unroll
        for (int s = 0; s < 2; ++s) {
            const int cbp = wave * 32 + s * 16;          // pair base within chunk
            union { uint4 u[2]; v16bf v; } a, b;
            // A: VGPR0-3 = pairs cbp+4*half.., VGPR4-7 = pairs cbp+8+4*half..
            a.u[0] = *(const uint4*)&bufA[j * ROWSTRIDE + cbp + 4 * half];
            a.u[1] = *(const uint4*)&bufA[j * ROWSTRIDE + cbp + 8 + 4 * half];
            // B: VGPR0-3 = pairs cbp+4*half.., VGPR4-7 = pairs cbp+8+4*half.. (same as A)
            const unsigned* wp = wpk + (size_t)j * WPK_ROW + c * (KC / 2) + cbp + 4 * half;
            b.u[0] = *(const uint4*)wp;
            b.u[1] = *(const uint4*)(wp + 8);
            acc = __builtin_amdgcn_wmma_f32_16x16x32_bf16(
                false, a.v, false, b.v, (short)0, acc, false, false);
        }
        // ---- pack + store next chunk into the other buffer ----
        if (more) {
            unsigned* bufN = sA[(c + 1) & 1];
            #pragma unroll
            for (int it = 0; it < 8; ++it) {
                const int col = it * 32 + sub * 4;
                const f4 vx = rx[it];
                ssq_local += vx.x * vx.x + vx.y * vx.y + vx.z * vx.z + vx.w * vx.w;
                const int p = col >> 1;
                bufN[row * ROWSTRIDE + p]     = pack_bf16(vx.x, vx.y);
                bufN[row * ROWSTRIDE + p + 1] = pack_bf16(vx.z, vx.w);
            }
        }
        __syncthreads();
    }

    // ---- reduce sum-of-squares and partial tiles through LDS ----
    atomicAdd(&ssq[row], ssq_local);
    #pragma unroll
    for (int v = 0; v < 8; ++v) {
        const int t = v + 8 * half;                      // C layout: M = vgpr + 8*half
        accb[wave * 256 + t * 16 + j] = acc[v];
    }
    __syncthreads();

    // ---- per-token epilogue: scale, bias, Sinkhorn-Knopp, store ----
    if (tid < 16) {
        const int t = tid;
        const float scale = rsqrtf(ssq[t] * (1.0f / D_EMBD) + 1.1920928955078125e-07f);
        const float alpha = alpha_p[0];
        float M[16];
        #pragma unroll
        for (int f = 0; f < 16; ++f) {
            float s = accb[        t * 16 + f] + accb[256 + t * 16 + f]
                    + accb[512 +   t * 16 + f] + accb[768 + t * 16 + f];
            M[f] = __expf(alpha * scale * s + bias[f]);
        }
        for (int it = 0; it < 20; ++it) {
            #pragma unroll
            for (int jj = 0; jj < 4; ++jj) {             // column normalize
                float cs = M[jj] + M[4 + jj] + M[8 + jj] + M[12 + jj];
                float r = 1.0f / (cs + 1e-8f);
                M[jj] *= r; M[4 + jj] *= r; M[8 + jj] *= r; M[12 + jj] *= r;
            }
            #pragma unroll
            for (int i = 0; i < 4; ++i) {                // row normalize
                float rs = M[i * 4] + M[i * 4 + 1] + M[i * 4 + 2] + M[i * 4 + 3];
                float r = 1.0f / (rs + 1e-8f);
                M[i * 4] *= r; M[i * 4 + 1] *= r; M[i * 4 + 2] *= r; M[i * 4 + 3] *= r;
            }
        }
        float* o = out + (size_t)(tok0 + t) * 16;
        #pragma unroll
        for (int q = 0; q < 4; ++q) {
            f4 v; v.x = M[q * 4]; v.y = M[q * 4 + 1]; v.z = M[q * 4 + 2]; v.w = M[q * 4 + 3];
            *(f4*)(o + q * 4) = v;
        }
    }
}

extern "C" void kernel_launch(void* const* d_in, const int* in_sizes, int n_in,
                              void* d_out, int out_size, void* d_ws, size_t ws_size,
                              hipStream_t stream) {
    const float* x     = (const float*)d_in[0];
    const float* W     = (const float*)d_in[1];
    const float* bias  = (const float*)d_in[2];
    const float* alpha = (const float*)d_in[3];
    float* out = (float*)d_out;
    unsigned* wpk = (unsigned*)d_ws;                 // needs 16*2048*4 = 128 KB scratch

    // 1) pack W to bf16 once (deterministic, every call)
    pack_w_kernel<<<16, NTHREADS, 0, stream>>>(W, wpk);

    // 2) main fused kernel
    const int tokens = in_sizes[0] / D_EMBD;         // B*T = 16384
    dim3 grid(tokens / TOK_PER_WG), block(NTHREADS);
    rsm_kernel<<<grid, block, 0, stream>>>(x, wpk, bias, alpha, out);
}
